// RoPEAttention_2774548873379
// MI455X (gfx1250) — compile-verified, matrix operand order checked by reading, not run on hardware
//
#include <hip/hip_runtime.h>

// ---------------------------------------------------------------------------
// RoPE causal attention for MI455X (gfx1250, wave32, WMMA + TDM).
// All GEMMs: bf16 operands, fp32 accumulation via v_wmma_f32_16x16x32_bf16.
// Block-cooperative GEMM: 8 waves share a 128x64 macro-tile; the 64x32 B tile
// is staged into LDS by the Tensor Data Mover (tensor_load_to_lds), double-
// buffered (one barrier per K-step) so the DMA for step k+32 overlaps the
// WMMAs of step k; the A fragment is register double-buffered the same way.
// Two-pass attention: the 134MB score tensor fits in the 192MB global L2.
// ---------------------------------------------------------------------------

#define B_SZ 4
#define L_SZ 2048
#define D_SZ 2048
#define H_SZ 2
#define DH_SZ (D_SZ / H_SZ)     // 1024
#define DQKV  (3 * D_SZ)        // 6144
#define NEG_BIG (-1.0e9f)

typedef __bf16 bf16;
typedef __attribute__((ext_vector_type(16))) __bf16 v16bf;
typedef __attribute__((ext_vector_type(8)))  __bf16 v8bf;
typedef __attribute__((ext_vector_type(8)))  float  v8f;
typedef __attribute__((ext_vector_type(4)))  unsigned int v4u;
typedef __attribute__((ext_vector_type(8)))  int v8i;
typedef __attribute__((ext_vector_type(4)))  int v4i;
typedef __attribute__((address_space(3))) bf16 lds_bf16;

union Frag16 { v16bf v; v8bf h[2]; };

__device__ __forceinline__ bf16 f2bf(float f) {
  unsigned u = __float_as_uint(f);
  u += 0x7FFFu + ((u >> 16) & 1u);          // round-to-nearest-even
  unsigned short hs = (unsigned short)(u >> 16);
  bf16 r;
  __builtin_memcpy(&r, &hs, sizeof(r));
  return r;
}
__device__ __forceinline__ float bf2f(bf16 b) {
  unsigned short hs;
  __builtin_memcpy(&hs, &b, sizeof(hs));
  return __uint_as_float(((unsigned)hs) << 16);
}

// CDNA5 16-bit A/B fragment layout: lanes 0-15 hold K {kk..kk+7, kk+16..kk+23},
// lanes 16-31 hold K {kk+8..kk+15, kk+24..kk+31} of their row/column.
__device__ __forceinline__ void load_frag(Frag16& f, const bf16* row, int kk, int half) {
  f.h[0] = *(const v8bf*)(row + kk + half * 8);
  f.h[1] = *(const v8bf*)(row + kk + 16 + half * 8);
}

// TDM: DMA a 2D bf16 tile (tile_d0 contiguous elems x tile_d1 rows, row stride
// in elements) from global memory into LDS. D# layout per CDNA5 ISA 8.3/8.4.
// This toolchain exposes the 6-arg builtin: (v4u, v8i, v4i, v4i, v8i, cpol).
__device__ __forceinline__ void tdm_load_2d(unsigned lds_addr, const bf16* gsrc,
                                            unsigned tile_d0, unsigned tile_d1,
                                            unsigned long long stride_elems) {
  const unsigned long long ga = (unsigned long long)gsrc;
  v4u g0;
  g0[0] = 1u;                                        // count=1, user descriptor
  g0[1] = lds_addr;                                  // lds byte address
  g0[2] = (unsigned)(ga & 0xFFFFFFFFu);              // global_addr[31:0]
  g0[3] = (unsigned)((ga >> 32) & 0x01FFFFFFu)       // global_addr[56:32]
        | (2u << 30);                                // type=2 ("image")
  v8i g1;
  const unsigned td0 = tile_d0;                      // tensor dims == tile dims
  const unsigned td1 = tile_d1;                      // (tile fully in-bounds)
  g1[0] = (int)(1u << 16);                           // data_size=1 (2B), mask=0
  g1[1] = (int)((td0 & 0xFFFFu) << 16);              // tensor_dim0[15:0]
  g1[2] = (int)(((td0 >> 16) & 0xFFFFu) | ((td1 & 0xFFFFu) << 16));
  g1[3] = (int)(((td1 >> 16) & 0xFFFFu) | ((tile_d0 & 0xFFFFu) << 16)); // tile_dim0
  g1[4] = (int)(tile_d1 & 0xFFFFu);                  // tile_dim1, tile_dim2=0 (2D)
  g1[5] = (int)(unsigned)(stride_elems & 0xFFFFFFFFu); // tensor_dim0_stride[31:0]
  g1[6] = (int)((unsigned)((stride_elems >> 32) & 0xFFFFu)); // stride0[47:32]
  g1[7] = 0;
  const v4i z4 = {0, 0, 0, 0};                       // groups 2/3 unused (2D)
  const v8i z8 = {0, 0, 0, 0, 0, 0, 0, 0};
  __builtin_amdgcn_tensor_load_to_lds(g0, g1, z4, z4, z8, 0);
}

// Block-cooperative GEMM: 256 threads = 8 waves compute a 128(M) x 64(N) fp32
// macro-tile of A[M,K] @ B[N,K]^T. Wave w owns rows [mBase+16w, +16).
// B tile (64 rows x 32 K) staged in LDS via TDM (wave w DMAs rows 8w..8w+7).
// Double-buffered LDS + register-double-buffered A; single barrier per step:
// the TDM issued in step i overwrites the buffer last read in step i-1, and
// the barrier at the top of step i orders those (each wave's ds reads complete
// before its WMMAs, which precede the barrier).
// Btile must point to __shared__ bf16 [2 * 64 * 32].
__device__ __forceinline__ void block_gemm_128x64(
    const bf16* __restrict__ A, size_t lda, int mBase,
    const bf16* __restrict__ Bw, size_t ldb, int nBase,
    int K, bf16* Btile, v8f acc[4]) {
  const int lane = threadIdx.x & 31;
  const int wib  = threadIdx.x >> 5;                 // wave-in-block 0..7
  const int half = lane >> 4;
  const int r    = lane & 15;
  const unsigned ldsBase =
      (unsigned)(unsigned long long)(lds_bf16*)Btile;
  const unsigned lds0 = ldsBase + (unsigned)(wib * 8 * 32 * sizeof(bf16));
  const unsigned lds1 = lds0 + 64 * 32 * sizeof(bf16);
  const bf16* tile0 = Btile;
  const bf16* tile1 = Btile + 64 * 32;
  const bf16* arow = A + (size_t)(mBase + wib * 16 + r) * lda;
  const bf16* bsrc = Bw + (size_t)(nBase + wib * 8) * ldb;   // this wave's rows

  tdm_load_2d(lds0, bsrc, 32, 8, ldb);               // prime LDS buffer 0
  Frag16 a_cur;
  load_frag(a_cur, arow, 0, half);                   // prime A fragment
  int p = 0;
  for (int kk = 0; kk < K; kk += 32) {
    __builtin_amdgcn_s_wait_tensorcnt(0);            // my DMA slice landed
    __syncthreads();            // buf[p] fully staged; prior buf[p^1] reads done
    const bool more = (kk + 32 < K);
    if (more)                                        // DMA next step's B tile
      tdm_load_2d(p ? lds0 : lds1, bsrc + kk + 32, 32, 8, ldb);
    __builtin_prefetch(arow + kk + 256, 0, 1);       // global_prefetch_b8
    Frag16 a_next;
    if (more) load_frag(a_next, arow, kk + 32, half); // overlaps WMMAs below
    const bf16* cur = p ? tile1 : tile0;
    // Issue all 8 ds_load_b128 first so LDS latency overlaps the WMMAs.
    Frag16 b0, b1, b2, b3;
    b0.h[0] = *(const v8bf*)(cur + (0 * 16 + r) * 32 + half * 8);
    b0.h[1] = *(const v8bf*)(cur + (0 * 16 + r) * 32 + 16 + half * 8);
    b1.h[0] = *(const v8bf*)(cur + (1 * 16 + r) * 32 + half * 8);
    b1.h[1] = *(const v8bf*)(cur + (1 * 16 + r) * 32 + 16 + half * 8);
    b2.h[0] = *(const v8bf*)(cur + (2 * 16 + r) * 32 + half * 8);
    b2.h[1] = *(const v8bf*)(cur + (2 * 16 + r) * 32 + 16 + half * 8);
    b3.h[0] = *(const v8bf*)(cur + (3 * 16 + r) * 32 + half * 8);
    b3.h[1] = *(const v8bf*)(cur + (3 * 16 + r) * 32 + 16 + half * 8);
    acc[0] = __builtin_amdgcn_wmma_f32_16x16x32_bf16(
        false, a_cur.v, false, b0.v, (short)0, acc[0], false, false);
    acc[1] = __builtin_amdgcn_wmma_f32_16x16x32_bf16(
        false, a_cur.v, false, b1.v, (short)0, acc[1], false, false);
    acc[2] = __builtin_amdgcn_wmma_f32_16x16x32_bf16(
        false, a_cur.v, false, b2.v, (short)0, acc[2], false, false);
    acc[3] = __builtin_amdgcn_wmma_f32_16x16x32_bf16(
        false, a_cur.v, false, b3.v, (short)0, acc[3], false, false);
    if (more) a_cur = a_next;
    p ^= 1;
  }
}

// ------------------------------ kernels ------------------------------------

__global__ __launch_bounds__(256)
void cvt_f32_bf16(const float* __restrict__ in, bf16* __restrict__ out, int n) {
  for (int i = blockIdx.x * blockDim.x + threadIdx.x; i < n;
       i += gridDim.x * blockDim.x)
    out[i] = f2bf(in[i]);
}

// qkv[m,n] = sum_k xb[m,k] * wqkvb[n,k] + bias[n]   (M=8192, N=6144, K=2048)
__global__ __launch_bounds__(256)
void qkv_gemm(const bf16* __restrict__ xb, const bf16* __restrict__ wb,
              const float* __restrict__ bias, bf16* __restrict__ qkv) {
  __shared__ __align__(128) bf16 Btile[2 * 64 * 32];
  const int nTiles = DQKV / 64;                      // 96
  const int mBase  = ((int)blockIdx.x / nTiles) * 128;
  const int nBase  = ((int)blockIdx.x % nTiles) * 64;
  v8f acc[4] = {};
  block_gemm_128x64(xb, D_SZ, mBase, wb, D_SZ, nBase, D_SZ, Btile, acc);
  const int lane = threadIdx.x & 31, wib = threadIdx.x >> 5;
  const int half = lane >> 4, r = lane & 15;
#pragma unroll
  for (int nt = 0; nt < 4; ++nt)
#pragma unroll
    for (int i = 0; i < 8; ++i) {
      const int row = mBase + wib * 16 + half * 8 + i;
      const int col = nBase + nt * 16 + r;
      qkv[(size_t)row * DQKV + col] = f2bf(acc[nt][i] + bias[col]);
    }
}

// RoPE(full head dim) on q,k; split heads; emit fp32 k/v outputs ([B,H,L,Dh])
// plus bf16 q_rot/k_rot ([B,H,L,Dh]) and bf16 V^T ([B,H,Dh,L]) for PV.
__global__ __launch_bounds__(256)
void rope_split(const bf16* __restrict__ qkv,
                bf16* __restrict__ qrot, bf16* __restrict__ krot,
                bf16* __restrict__ vT,
                float* __restrict__ k_out, float* __restrict__ v_out) {
  const int bl = blockIdx.x;             // 0 .. B*L-1
  const int b  = bl / L_SZ, l = bl % L_SZ;
  const int t  = threadIdx.x;            // 256 threads
  const bf16* row = qkv + (size_t)bl * DQKV;
  const float LOG_BASE = 9.2103403719761836f;      // ln(10000)
#pragma unroll
  for (int h = 0; h < H_SZ; ++h) {
    const size_t headRow = ((size_t)(b * H_SZ + h) * L_SZ + l) * DH_SZ;
    const size_t vtBase  = (size_t)(b * H_SZ + h) * DH_SZ * L_SZ;
#pragma unroll
    for (int j = 0; j < 2; ++j) {
      const int i = t + j * 256;                   // rope pair 0..511
      const float inv   = __expf(-LOG_BASE * ((float)i * (1.0f / 512.0f)));
      const float theta = (float)l * inv;
      float s, c; __sincosf(theta, &s, &c);
      // q
      const float q1 = bf2f(row[h * DH_SZ + i]);
      const float q2 = bf2f(row[h * DH_SZ + 512 + i]);
      qrot[headRow + i]       = f2bf(q1 * c - q2 * s);
      qrot[headRow + 512 + i] = f2bf(q1 * s + q2 * c);
      // k (also fp32 output)
      const float k1 = bf2f(row[D_SZ + h * DH_SZ + i]);
      const float k2 = bf2f(row[D_SZ + h * DH_SZ + 512 + i]);
      const float r1 = k1 * c - k2 * s, r2 = k1 * s + k2 * c;
      krot[headRow + i]       = f2bf(r1);
      krot[headRow + 512 + i] = f2bf(r2);
      k_out[headRow + i]       = r1;
      k_out[headRow + 512 + i] = r2;
      // v: passthrough fp32 output + bf16 transpose for PV B-operand
      const float v1 = bf2f(row[2 * D_SZ + h * DH_SZ + i]);
      const float v2 = bf2f(row[2 * D_SZ + h * DH_SZ + 512 + i]);
      v_out[headRow + i]       = v1;
      v_out[headRow + 512 + i] = v2;
      vT[vtBase + (size_t)i * L_SZ + l]         = f2bf(v1);
      vT[vtBase + (size_t)(i + 512) * L_SZ + l] = f2bf(v2);
    }
  }
}

// scores[bh, qi, ki] = scale * q.k + causal_mask   (per head: 2048x2048, K=1024)
__global__ __launch_bounds__(256)
void attn_scores(const bf16* __restrict__ q, const bf16* __restrict__ k,
                 float* __restrict__ scores) {
  __shared__ __align__(128) bf16 Btile[2 * 64 * 32];
  const int tilesPerBH = (L_SZ / 128) * (L_SZ / 64);  // 512
  const int bh    = (int)blockIdx.x / tilesPerBH;
  const int rem   = (int)blockIdx.x % tilesPerBH;
  const int mBase = (rem / (L_SZ / 64)) * 128;
  const int nBase = (rem % (L_SZ / 64)) * 64;
  float* sh = scores + (size_t)bh * L_SZ * L_SZ;
  const int lane = threadIdx.x & 31, wib = threadIdx.x >> 5;
  const int half = lane >> 4, r = lane & 15;

  if (nBase > mBase + 127) {                // macro-tile fully above diagonal
#pragma unroll
    for (int nt = 0; nt < 4; ++nt)
#pragma unroll
      for (int i = 0; i < 8; ++i) {
        const int row = mBase + wib * 16 + half * 8 + i;
        const int col = nBase + nt * 16 + r;
        sh[(size_t)row * L_SZ + col] = NEG_BIG;
      }
    return;
  }
  const bf16* qh = q + (size_t)bh * L_SZ * DH_SZ;
  const bf16* kh = k + (size_t)bh * L_SZ * DH_SZ;
  v8f acc[4] = {};
  block_gemm_128x64(qh, DH_SZ, mBase, kh, DH_SZ, nBase, DH_SZ, Btile, acc);
  const float scale = 0.03125f;             // 1/sqrt(1024)
#pragma unroll
  for (int nt = 0; nt < 4; ++nt)
#pragma unroll
    for (int i = 0; i < 8; ++i) {
      const int row = mBase + wib * 16 + half * 8 + i;
      const int col = nBase + nt * 16 + r;
      const float v = acc[nt][i] * scale + (col <= row ? 0.0f : NEG_BIG);
      sh[(size_t)row * L_SZ + col] = v;
    }
}

// Row-wise softmax over 2048 fp32 scores -> bf16 probs (LDS tree reductions).
__global__ __launch_bounds__(256)
void attn_softmax(const float* __restrict__ scores, bf16* __restrict__ probs) {
  __shared__ float red[256];
  const int row = blockIdx.x;
  const int t   = threadIdx.x;
  const float* src = scores + (size_t)row * L_SZ;
  bf16* dst = probs + (size_t)row * L_SZ;
  float v[8];
  float m = -3.0e38f;
#pragma unroll
  for (int j = 0; j < 8; ++j) { v[j] = src[t + j * 256]; m = fmaxf(m, v[j]); }
  red[t] = m; __syncthreads();
  for (int s = 128; s > 0; s >>= 1) {
    if (t < s) red[t] = fmaxf(red[t], red[t + s]);
    __syncthreads();
  }
  m = red[0]; __syncthreads();
  float sum = 0.0f;
#pragma unroll
  for (int j = 0; j < 8; ++j) { v[j] = __expf(v[j] - m); sum += v[j]; }
  red[t] = sum; __syncthreads();
  for (int s = 128; s > 0; s >>= 1) {
    if (t < s) red[t] += red[t + s];
    __syncthreads();
  }
  const float inv = 1.0f / red[0];
#pragma unroll
  for (int j = 0; j < 8; ++j) dst[t + j * 256] = f2bf(v[j] * inv);
}

// vh[b, qi, h*Dh+dh] = sum_ki probs[bh,qi,ki] * vT[bh,dh,ki]
// Causality: probs[qi, ki>qi] == 0, so the K loop stops past the macro-tile's
// diagonal (mBase+128, already a multiple of 32).
__global__ __launch_bounds__(256)
void attn_pv(const bf16* __restrict__ probs, const bf16* __restrict__ vT,
             bf16* __restrict__ vh) {
  __shared__ __align__(128) bf16 Btile[2 * 64 * 32];
  const int tilesPerBH = (L_SZ / 128) * (DH_SZ / 64); // 256
  const int bh    = (int)blockIdx.x / tilesPerBH;
  const int rem   = (int)blockIdx.x % tilesPerBH;
  const int mBase = (rem / (DH_SZ / 64)) * 128;
  const int nBase = (rem % (DH_SZ / 64)) * 64;
  const bf16* ph = probs + (size_t)bh * L_SZ * L_SZ;
  const bf16* vb = vT + (size_t)bh * DH_SZ * L_SZ;
  const int Klim = (mBase + 128) < L_SZ ? (mBase + 128) : L_SZ;
  v8f acc[4] = {};
  block_gemm_128x64(ph, L_SZ, mBase, vb, L_SZ, nBase, Klim, Btile, acc);
  const int b = bh / H_SZ, h = bh % H_SZ;
  const int lane = threadIdx.x & 31, wib = threadIdx.x >> 5;
  const int half = lane >> 4, r = lane & 15;
#pragma unroll
  for (int nt = 0; nt < 4; ++nt)
#pragma unroll
    for (int i = 0; i < 8; ++i) {
      const int row = mBase + wib * 16 + half * 8 + i;   // query position
      const int col = nBase + nt * 16 + r;               // dh
      vh[((size_t)(b * L_SZ + row)) * D_SZ + h * DH_SZ + col] = f2bf(acc[nt][i]);
    }
}

// out[m,n] = sum_k vh[m,k] * wo[n,k] + bias[n]   (fp32 output)
__global__ __launch_bounds__(256)
void out_proj(const bf16* __restrict__ vh, const bf16* __restrict__ wo,
              const float* __restrict__ bias, float* __restrict__ out) {
  __shared__ __align__(128) bf16 Btile[2 * 64 * 32];
  const int nTiles = D_SZ / 64;                      // 32
  const int mBase  = ((int)blockIdx.x / nTiles) * 128;
  const int nBase  = ((int)blockIdx.x % nTiles) * 64;
  v8f acc[4] = {};
  block_gemm_128x64(vh, D_SZ, mBase, wo, D_SZ, nBase, D_SZ, Btile, acc);
  const int lane = threadIdx.x & 31, wib = threadIdx.x >> 5;
  const int half = lane >> 4, r = lane & 15;
#pragma unroll
  for (int nt = 0; nt < 4; ++nt)
#pragma unroll
    for (int i = 0; i < 8; ++i) {
      const int row = mBase + wib * 16 + half * 8 + i;
      const int col = nBase + nt * 16 + r;
      out[(size_t)row * D_SZ + col] = acc[nt][i] + bias[col];
    }
}

// ------------------------------ launcher -----------------------------------

extern "C" void kernel_launch(void* const* d_in, const int* in_sizes, int n_in,
                              void* d_out, int out_size, void* d_ws, size_t ws_size,
                              hipStream_t stream) {
  const float* x      = (const float*)d_in[0];
  // d_in[1] = mask : unused, causal mask is applied analytically
  const float* Wqkv_w = (const float*)d_in[2];
  const float* Wqkv_b = (const float*)d_in[3];
  const float* out_w  = (const float*)d_in[4];
  const float* out_b  = (const float*)d_in[5];

  float* out   = (float*)d_out;                                 // [B,L,D]
  float* k_out = out + (size_t)B_SZ * L_SZ * D_SZ;              // [B,H,L,Dh]
  float* v_out = k_out + (size_t)B_SZ * H_SZ * L_SZ * DH_SZ;    // [B,H,L,Dh]

  char* w = (char*)d_ws;
  size_t off = 0;
  auto take = [&](size_t bytes) -> void* {
    void* p = (void*)(w + off);
    off = (off + bytes + 255) & ~(size_t)255;
    return p;
  };
  bf16*  xb     = (bf16*)take((size_t)B_SZ * L_SZ * D_SZ * sizeof(bf16));
  bf16*  wqkvb  = (bf16*)take((size_t)DQKV * D_SZ * sizeof(bf16));
  bf16*  outwb  = (bf16*)take((size_t)D_SZ * D_SZ * sizeof(bf16));
  bf16*  qkvb   = (bf16*)take((size_t)B_SZ * L_SZ * DQKV * sizeof(bf16));
  bf16*  qrot   = (bf16*)take((size_t)B_SZ * H_SZ * L_SZ * DH_SZ * sizeof(bf16));
  bf16*  krot   = (bf16*)take((size_t)B_SZ * H_SZ * L_SZ * DH_SZ * sizeof(bf16));
  bf16*  vT     = (bf16*)take((size_t)B_SZ * H_SZ * DH_SZ * L_SZ * sizeof(bf16));
  float* scores = (float*)take((size_t)B_SZ * H_SZ * L_SZ * L_SZ * sizeof(float));
  bf16*  probs  = qkvb;   // alias: qkv bf16 fully consumed by rope_split
  bf16*  vh     = xb;     // alias: x bf16 fully consumed by qkv_gemm

  dim3 blk(256);
  // 1. fp32 -> bf16 conversions
  cvt_f32_bf16<<<4096, blk, 0, stream>>>(x, xb, B_SZ * L_SZ * D_SZ);
  cvt_f32_bf16<<<4096, blk, 0, stream>>>(Wqkv_w, wqkvb, DQKV * D_SZ);
  cvt_f32_bf16<<<2048, blk, 0, stream>>>(out_w, outwb, D_SZ * D_SZ);
  // 2. QKV projection (WMMA + TDM-staged B)
  qkv_gemm<<<(B_SZ * L_SZ / 128) * (DQKV / 64), blk, 0, stream>>>(
      xb, wqkvb, Wqkv_b, qkvb);
  // 3. RoPE + head split + fp32 k/v outputs + V^T
  rope_split<<<B_SZ * L_SZ, blk, 0, stream>>>(qkvb, qrot, krot, vT, k_out, v_out);
  // 4. scores (WMMA, causal macro-tile skip)
  attn_scores<<<B_SZ * H_SZ * (L_SZ / 128) * (L_SZ / 64), blk, 0, stream>>>(
      qrot, krot, scores);
  // 5. softmax (LDS reductions)
  attn_softmax<<<B_SZ * H_SZ * L_SZ, blk, 0, stream>>>(scores, probs);
  // 6. PV (WMMA, diagonal-limited K loop)
  attn_pv<<<B_SZ * H_SZ * (L_SZ / 128) * (DH_SZ / 64), blk, 0, stream>>>(
      probs, vT, vh);
  // 7. output projection (WMMA, fp32 out + bias)
  out_proj<<<(B_SZ * L_SZ / 128) * (D_SZ / 64), blk, 0, stream>>>(
      vh, outwb, out_b, out);
}
